// FastMultipleNegativesRankingLoss_41841571398368
// MI455X (gfx1250) — matrix operand order repaired and checked by reading, not compiled, not run on hardware
//
#include <hip/hip_runtime.h>

// FastMultipleNegativesRankingLoss for MI455X (gfx1250, wave32, WMMA).
//
// loss = mean_i [ logsumexp_j(20*cos(a_i,b_j)) - 20*cos(a_i,b_i) ]
//
// Compute-bound (275 GFLOP vs 96 MB) -> bf16 WMMA with fp32 accumulate.
// Cosines bounded in [-1,1] => lse = 20 + log(sum exp(s-20)) with a FIXED
// shift: no online max needed.
// 320KB LDS: whole 128x1024 bf16 A-strip staged once per block (264KB) +
// streaming 128x128 B tile (35KB). bf16 B (32MB) lives in the 192MB L2.
// Tile staging uses gfx1250 GLOBAL_LOAD_ASYNC_TO_LDS_B128 (ASYNCcnt),
// bypassing VGPRs; fragments are preloaded in register arrays so the
// scheduler can overlap ds_load latency with earlier WMMAs.

typedef unsigned short u16;
typedef unsigned int   u32;
typedef __attribute__((ext_vector_type(16))) __bf16 v16bf;
typedef __attribute__((ext_vector_type(8)))  float  v8f;

#define DIM    1024
#define LSCALE 20.0f

__device__ __forceinline__ u16 f2bf(float f) {
    u32 u = __float_as_uint(f);
    u32 r = (u + 0x7FFFu + ((u >> 16) & 1u)) >> 16;   // round-to-nearest-even
    return (u16)r;
}
__device__ __forceinline__ float bf2f(u16 h) {
    return __uint_as_float(((u32)h) << 16);
}

// gfx1250 async copy: 16B per lane, global (SADDR base + u32 VGPR offset) -> LDS.
// ldsoff = byte offset into workgroup LDS (low 32 bits of generic shared ptr).
__device__ __forceinline__ void async_g2l_b128(u32 ldsoff, u32 goff, const void* base) {
    asm volatile("global_load_async_to_lds_b128 %0, %1, %2"
                 :: "v"(ldsoff), "v"(goff), "s"(base) : "memory");
}
__device__ __forceinline__ void wait_async0() {
    asm volatile("s_wait_asynccnt 0" ::: "memory");
}

// ---------------------------------------------------------------------------
// Kernel 1: L2-normalize one fp32 row (1024 elems) -> bf16 row.
// ---------------------------------------------------------------------------
__global__ __launch_bounds__(256)
void norm_bf16_kernel(const float* __restrict__ in, u16* __restrict__ out) {
    const int row = blockIdx.x;
    const int tid = threadIdx.x;
    const float4 v = ((const float4*)(in + (size_t)row * DIM))[tid];
    float ss = v.x * v.x + v.y * v.y + v.z * v.z + v.w * v.w;
    #pragma unroll
    for (int m = 16; m >= 1; m >>= 1) ss += __shfl_xor(ss, m, 32);
    __shared__ float red[8];
    if ((tid & 31) == 0) red[tid >> 5] = ss;
    __syncthreads();
    float tot = 0.f;
    #pragma unroll
    for (int w = 0; w < 8; ++w) tot += red[w];
    const float r = rsqrtf(tot);
    const u32 p0 = (u32)f2bf(v.x * r) | ((u32)f2bf(v.y * r) << 16);
    const u32 p1 = (u32)f2bf(v.z * r) | ((u32)f2bf(v.w * r) << 16);
    ((uint2*)(out + (size_t)row * DIM))[tid] = make_uint2(p0, p1);
}

// ---------------------------------------------------------------------------
// Kernel 2: pos[i] = 20 * dot(a_n[i], b_n[i]) on the bf16 normalized rows.
// ---------------------------------------------------------------------------
__global__ __launch_bounds__(256)
void pos_kernel(const u16* __restrict__ a, const u16* __restrict__ b,
                float* __restrict__ pos) {
    const int row = blockIdx.x;
    const int tid = threadIdx.x;
    const uint2 av = ((const uint2*)(a + (size_t)row * DIM))[tid];
    const uint2 bv = ((const uint2*)(b + (size_t)row * DIM))[tid];
    float d = bf2f((u16)av.x)         * bf2f((u16)bv.x)
            + bf2f((u16)(av.x >> 16)) * bf2f((u16)(bv.x >> 16))
            + bf2f((u16)av.y)         * bf2f((u16)bv.y)
            + bf2f((u16)(av.y >> 16)) * bf2f((u16)(bv.y >> 16));
    #pragma unroll
    for (int m = 16; m >= 1; m >>= 1) d += __shfl_xor(d, m, 32);
    __shared__ float red[8];
    if ((tid & 31) == 0) red[tid >> 5] = d;
    __syncthreads();
    if (tid == 0) {
        float tot = 0.f;
        #pragma unroll
        for (int w = 0; w < 8; ++w) tot += red[w];
        pos[row] = LSCALE * tot;
    }
}

// ---------------------------------------------------------------------------
// Kernel 3: 128-row strip of A vs all Bb columns; WMMA bf16 GEMM with
// fixed-shift sum-of-exp; writes lse[row] = 20 + log(sumexp).
// Block = 256 thr = 8 waves; wave w owns rows [w*16, w*16+16).
// ---------------------------------------------------------------------------
__global__ __launch_bounds__(256, 1)
void sim_lse_kernel(const u16* __restrict__ A, const u16* __restrict__ B,
                    float* __restrict__ lse, const int Bb) {
    __shared__ u16 lds_a[128][DIM + 8];   // 128x1024 bf16 strip, padded: 264192 B
    __shared__ u16 lds_b[128][128 + 8];   // 128(N) x 128(K) tile, padded: 34816 B

    const int tid  = threadIdx.x;
    const int wv   = tid >> 5;
    const int lane = tid & 31;
    const int hlf  = lane >> 4;       // lane half selects K sub-range (ISA layout)
    const int l16  = lane & 15;       // A-frag: M index; B-frag: N index
    const long mblk = (long)blockIdx.x * 128;

    // ---- Stage whole A strip into LDS via async copies (global read once) ----
    for (int i = tid; i < 128 * (DIM / 8); i += 256) {
        const int row = i >> 7;               // DIM/8 = 128 16B-chunks per row
        const int c8  = i & 127;
        const u32 ldsoff = (u32)(uintptr_t)&lds_a[row][c8 * 8];
        const u32 goff   = (u32)(((mblk + row) * DIM + c8 * 8) * sizeof(u16));
        async_g2l_b128(ldsoff, goff, A);
    }
    wait_async0();
    __syncthreads();

    float sumacc[8] = {};   // per-lane running sum of exp(s-20), one per C-VGPR row

    for (int n0 = 0; n0 < Bb; n0 += 128) {
        v8f acc[8] = {};    // 8 x (16x16 f32) = wave tile 16(M) x 128(N)

        for (int kc = 0; kc < DIM; kc += 128) {
            __syncthreads();                   // previous B tile fully consumed
            // async-stage B tile: rows n0..n0+127 (output cols), K = kc..kc+127
            #pragma unroll
            for (int i = 0; i < 8; ++i) {      // 2048 chunks / 256 threads
                const int idx = tid + i * 256;
                const int row = idx >> 4;
                const int c8  = idx & 15;
                const u32 ldsoff = (u32)(uintptr_t)&lds_b[row][c8 * 8];
                const u32 goff =
                    (u32)((((long)(n0 + row)) * DIM + kc + c8 * 8) * sizeof(u16));
                async_g2l_b128(ldsoff, goff, B);
            }
            wait_async0();
            __syncthreads();

            #pragma unroll
            for (int k = 0; k < 128; k += 32) {
                // A fragment 16x32: lane half splits K {0-7,16-23}/{8-15,24-31}
                union { uint4 u[2]; v16bf v; } af;
                af.u[0] = *(const uint4*)&lds_a[wv * 16 + l16][kc + k + 8 * hlf];
                af.u[1] = *(const uint4*)&lds_a[wv * 16 + l16][kc + k + 16 + 8 * hlf];
                // Preload ALL 8 B fragments so the scheduler can overlap
                // ds_load latency with earlier WMMAs (partial s_wait_dscnt).
                union { uint4 u[2]; v16bf v; } bfr[8];
                #pragma unroll
                for (int t = 0; t < 8; ++t) {
                    bfr[t].u[0] = *(const uint4*)&lds_b[t * 16 + l16][k + 8 * hlf];
                    bfr[t].u[1] = *(const uint4*)&lds_b[t * 16 + l16][k + 16 + 8 * hlf];
                }
                #pragma unroll
                for (int t = 0; t < 8; ++t)
                    acc[t] = __builtin_amdgcn_wmma_f32_16x16x32_bf16(
                        false, af.v, false, bfr[t].v, (short)0, acc[t], false, false);
            }
        }
        // acc holds cosines; fixed-shift sum of exp (logits bounded by +/-20)
        #pragma unroll
        for (int t = 0; t < 8; ++t) {
            #pragma unroll
            for (int r = 0; r < 8; ++r)
                sumacc[r] += __expf(LSCALE * acc[t][r] - LSCALE);
        }
    }

    // C-layout: lane L holds column N=L%16 for rows r + 8*(L>=16);
    // reduce each row across its 16-lane group, then lanes 0 / 16 write.
    #pragma unroll
    for (int r = 0; r < 8; ++r) {
        float s = sumacc[r];
        #pragma unroll
        for (int m = 1; m < 16; m <<= 1) s += __shfl_xor(s, m, 32);
        if (l16 == 0)
            lse[mblk + wv * 16 + 8 * hlf + r] = LSCALE + __logf(s);
    }
}

// ---------------------------------------------------------------------------
// Kernel 4: loss = mean(lse - pos) over Ba rows -> d_out[0].
// ---------------------------------------------------------------------------
__global__ __launch_bounds__(256)
void loss_kernel(const float* __restrict__ lse, const float* __restrict__ pos,
                 float* __restrict__ out, const int Ba) {
    const int tid = threadIdx.x;
    float s = 0.f;
    for (int i = tid; i < Ba; i += 256) s += lse[i] - pos[i];
    #pragma unroll
    for (int m = 16; m >= 1; m >>= 1) s += __shfl_xor(s, m, 32);
    __shared__ float red[8];
    if ((tid & 31) == 0) red[tid >> 5] = s;
    __syncthreads();
    if (tid == 0) {
        float tot = 0.f;
        #pragma unroll
        for (int w = 0; w < 8; ++w) tot += red[w];
        out[0] = tot / (float)Ba;
    }
}

// ---------------------------------------------------------------------------
extern "C" void kernel_launch(void* const* d_in, const int* in_sizes, int n_in,
                              void* d_out, int out_size, void* d_ws, size_t ws_size,
                              hipStream_t stream) {
    (void)n_in; (void)out_size; (void)ws_size;
    const float* a = (const float*)d_in[0];
    const float* b = (const float*)d_in[1];
    const int Ba = in_sizes[0] / DIM;   // 8192
    const int Bb = in_sizes[1] / DIM;   // 16384

    // workspace layout (~48 MB): bf16 normalized copies + pos + lse
    char* ws = (char*)d_ws;
    u16*   a16 = (u16*)ws;   ws += (size_t)Ba * DIM * sizeof(u16);
    u16*   b16 = (u16*)ws;   ws += (size_t)Bb * DIM * sizeof(u16);
    float* pos = (float*)ws; ws += (size_t)Ba * sizeof(float);
    float* lse = (float*)ws;

    norm_bf16_kernel<<<Ba, 256, 0, stream>>>(a, a16);
    norm_bf16_kernel<<<Bb, 256, 0, stream>>>(b, b16);
    pos_kernel<<<Ba, 256, 0, stream>>>(a16, b16, pos);
    sim_lse_kernel<<<Ba / 128, 256, 0, stream>>>(a16, b16, lse, Bb);
    loss_kernel<<<1, 256, 0, stream>>>(lse, pos, (float*)d_out, Ba);
}
